// MoETorchFFN_2774548873700
// MI455X (gfx1250) — matrix operand order repaired and checked by reading, not compiled, not run on hardware
//
#include <hip/hip_runtime.h>

// ---------------------------------------------------------------------------
// MoE FFN (top-2 of 8 experts, SwiGLU) for MI455X / gfx1250.
// T=4096 tokens, D=1024, HID=2048, E=8, K=2.
// Strategy: bf16 WMMA (16x16x32, fp32 accum) with routed (gathered) GEMMs.
// ---------------------------------------------------------------------------

#define T_TOK 4096
#define DIM   1024
#define HID   2048
#define NEXP  8

typedef __attribute__((ext_vector_type(16))) __bf16 v16bf;
typedef __attribute__((ext_vector_type(8)))  float  v8f;

union Frag {
    v16bf v;
    uint4 q[2];
};

__device__ __forceinline__ unsigned short f2bf(float f) {
    unsigned int u = __float_as_uint(f);
    u += 0x7FFFu + ((u >> 16) & 1u);        // round-to-nearest-even
    return (unsigned short)(u >> 16);
}

// ---------------------------------------------------------------------------
// Small prep kernels
// ---------------------------------------------------------------------------

__global__ void init_counts(int* cnt) {
    if (threadIdx.x < NEXP) cnt[threadIdx.x] = 0;
}

__global__ void zero_f32(float* p, int n4) {
    int i = blockIdx.x * blockDim.x + threadIdx.x;
    if (i < n4) ((float4*)p)[i] = make_float4(0.f, 0.f, 0.f, 0.f);
}

__global__ void convert_x_bf16(const float* __restrict__ x,
                               unsigned short* __restrict__ xb, int n4) {
    int i = blockIdx.x * blockDim.x + threadIdx.x;
    if (i >= n4) return;
    float4 v = ((const float4*)x)[i];
    unsigned int lo = (unsigned int)f2bf(v.x) | ((unsigned int)f2bf(v.y) << 16);
    unsigned int hi = (unsigned int)f2bf(v.z) | ((unsigned int)f2bf(v.w) << 16);
    ((uint2*)xb)[i] = make_uint2(lo, hi);
}

// src [e][R][C] f32 (row-major) -> dst [e][C][R] bf16 (K-contiguous for WMMA B)
__global__ __launch_bounds__(256) void transpose_to_bf16(
    const float* __restrict__ src, unsigned short* __restrict__ dst,
    int R, int C) {
    __shared__ float tile[32][33];
    size_t eoff = (size_t)blockIdx.z * R * C;
    src += eoff;  dst += eoff;
    int c0 = blockIdx.x * 32, r0 = blockIdx.y * 32;
    int tx = threadIdx.x, ty = threadIdx.y;          // block (32,8)
#pragma unroll
    for (int k = 0; k < 4; k++)
        tile[ty + k * 8][tx] = src[(size_t)(r0 + ty + k * 8) * C + c0 + tx];
    __syncthreads();
#pragma unroll
    for (int k = 0; k < 4; k++)
        dst[(size_t)(c0 + ty + k * 8) * R + r0 + tx] = f2bf(tile[tx][ty + k * 8]);
}

// ---------------------------------------------------------------------------
// Gating: one wave32 per token. scores = x @ Wg, top-2, softmax over pair,
// append token to each chosen expert's list.
// ---------------------------------------------------------------------------
__global__ __launch_bounds__(256) void moe_gate(
    const float* __restrict__ x, const float* __restrict__ Wg,
    int* __restrict__ cnt, int* __restrict__ toklist, float* __restrict__ wlist) {
    int wid  = threadIdx.x >> 5;
    int lane = threadIdx.x & 31;
    int t = blockIdx.x * 8 + wid;
    if (t >= T_TOK) return;

    float s[NEXP];
#pragma unroll
    for (int e = 0; e < NEXP; e++) s[e] = 0.f;

    const float* xr = x + (size_t)t * DIM;
    for (int d = lane; d < DIM; d += 32) {
        float xv = xr[d];
        const float4* wg = (const float4*)(Wg + (size_t)d * NEXP);
        float4 w0 = wg[0], w1 = wg[1];
        s[0] += xv * w0.x; s[1] += xv * w0.y; s[2] += xv * w0.z; s[3] += xv * w0.w;
        s[4] += xv * w1.x; s[5] += xv * w1.y; s[6] += xv * w1.z; s[7] += xv * w1.w;
    }
#pragma unroll
    for (int off = 16; off > 0; off >>= 1)
#pragma unroll
        for (int e = 0; e < NEXP; e++) s[e] += __shfl_xor(s[e], off, 32);

    if (lane == 0) {
        int e0 = 0; float b0 = s[0];
#pragma unroll
        for (int e = 1; e < NEXP; e++) if (s[e] > b0) { b0 = s[e]; e0 = e; }
        int e1 = -1; float b1 = -3.4e38f;
#pragma unroll
        for (int e = 0; e < NEXP; e++)
            if (e != e0 && s[e] > b1) { b1 = s[e]; e1 = e; }
        float ex = __expf(b1 - b0);          // b1 <= b0
        float w0 = 1.f / (1.f + ex);
        float w1 = ex / (1.f + ex);
        int p0 = atomicAdd(&cnt[e0], 1);
        toklist[e0 * T_TOK + p0] = t;  wlist[e0 * T_TOK + p0] = w0;
        int p1 = atomicAdd(&cnt[e1], 1);
        toklist[e1 * T_TOK + p1] = t;  wlist[e1 * T_TOK + p1] = w1;
    }
}

// ---------------------------------------------------------------------------
// GEMM tiling shared by both matmul kernels.
// BM=128, BN=64, BK=32. 8 waves (4x2), wave owns 32x32 = 2x2 WMMA tiles.
// LDS row stride 48 ushorts (96B): 16B-aligned b128 fragment loads.
// ---------------------------------------------------------------------------
#define BM 128
#define BN 64
#define BK 32
#define LDST 48

__device__ __forceinline__ v8f wmma_bf16(const v16bf& a, const v16bf& b, v8f c) {
    return __builtin_amdgcn_wmma_f32_16x16x32_bf16(
        false, a, false, b, (short)0, c, false, false);
}

// GEMM1 + SwiGLU: for expert e, gathered rows:
//   h[r][n] = silu(x[tok]@W1[:,n]) * (x[tok]@W3[:,n])   stored compact bf16.
__global__ __launch_bounds__(256) void moe_gemm1(
    const unsigned short* __restrict__ xb,    // [T][DIM] bf16
    const unsigned short* __restrict__ W1t,   // [HID][DIM] bf16 (N-major)
    const unsigned short* __restrict__ W3t,   // [HID][DIM]
    unsigned short* __restrict__ hbuf,        // [T][HID] compact
    const int* __restrict__ cnt_e,
    const int* __restrict__ toklist_e) {
    int count = *cnt_e;
    int mbase = blockIdx.x * BM;
    if (mbase >= count) return;                 // routing sparsity: skip tile
    int n0 = blockIdx.y * BN;

    __shared__ unsigned short As[BM * LDST];
    __shared__ unsigned short B1s[BN * LDST];
    __shared__ unsigned short B3s[BN * LDST];
    __shared__ int rows[BM];

    int tid = threadIdx.x;
    if (tid < BM) {
        int g = mbase + tid;
        rows[tid] = (g < count) ? toklist_e[g] : 0;
    }
    __syncthreads();

    int wid = tid >> 5, lane = tid & 31;
    int waveM = (wid >> 1) * 32, waveN = (wid & 1) * 32;
    int lr = lane & 15, lh = lane >> 4;

    v8f zc = {0.f,0.f,0.f,0.f,0.f,0.f,0.f,0.f};
    v8f c1[2][2], c3[2][2];
#pragma unroll
    for (int i = 0; i < 2; i++)
#pragma unroll
        for (int j = 0; j < 2; j++) { c1[i][j] = zc; c3[i][j] = zc; }

    for (int kb = 0; kb < DIM; kb += BK) {
        // A tile: 128x32, 2 x b128 per thread, row indices via gather list
#pragma unroll
        for (int j = 0; j < 2; j++) {
            int chunk = tid + j * 256;
            int row = chunk >> 2, cc = (chunk & 3) * 8;
            uint4 v = *(const uint4*)(xb + (size_t)rows[row] * DIM + kb + cc);
            *(uint4*)(&As[row * LDST + cc]) = v;
        }
        // B tiles: 64x32 each, 1 x b128 per thread per matrix
        {
            int nn = tid >> 2, cc = (tid & 3) * 8;
            uint4 v1 = *(const uint4*)(W1t + (size_t)(n0 + nn) * DIM + kb + cc);
            uint4 v3 = *(const uint4*)(W3t + (size_t)(n0 + nn) * DIM + kb + cc);
            *(uint4*)(&B1s[nn * LDST + cc]) = v1;
            *(uint4*)(&B3s[nn * LDST + cc]) = v3;
        }
        __syncthreads();

        Frag a[2], b1[2], b3[2];
#pragma unroll
        for (int tm = 0; tm < 2; tm++) {
            int base = (waveM + tm * 16 + lr) * LDST + lh * 16;
            a[tm].q[0] = *(const uint4*)(&As[base]);
            a[tm].q[1] = *(const uint4*)(&As[base + 8]);
        }
#pragma unroll
        for (int tn = 0; tn < 2; tn++) {
            int base = (waveN + tn * 16 + lr) * LDST + lh * 16;
            b1[tn].q[0] = *(const uint4*)(&B1s[base]);
            b1[tn].q[1] = *(const uint4*)(&B1s[base + 8]);
            b3[tn].q[0] = *(const uint4*)(&B3s[base]);
            b3[tn].q[1] = *(const uint4*)(&B3s[base + 8]);
        }
#pragma unroll
        for (int tm = 0; tm < 2; tm++)
#pragma unroll
            for (int tn = 0; tn < 2; tn++) {
                c1[tm][tn] = wmma_bf16(a[tm].v, b1[tn].v, c1[tm][tn]);
                c3[tm][tn] = wmma_bf16(a[tm].v, b3[tn].v, c3[tm][tn]);
            }
        __syncthreads();
    }

    // SwiGLU epilogue -> compact bf16 h
#pragma unroll
    for (int tm = 0; tm < 2; tm++)
#pragma unroll
        for (int tn = 0; tn < 2; tn++)
#pragma unroll
            for (int j = 0; j < 8; j++) {
                float v1 = c1[tm][tn][j];
                float v3 = c3[tm][tn][j];
                float hv = (v1 / (1.f + __expf(-v1))) * v3;
                int m = mbase + waveM + tm * 16 + j + 8 * lh;
                int n = n0 + waveN + tn * 16 + lr;
                hbuf[(size_t)m * HID + n] = f2bf(hv);
            }
}

// GEMM2: out[tok] += w * (h_compact @ W2t), scatter via token list.
__global__ __launch_bounds__(256) void moe_gemm2(
    const unsigned short* __restrict__ hbuf,  // [T][HID] compact bf16
    const unsigned short* __restrict__ W2t,   // [DIM][HID] bf16 (N-major)
    float* __restrict__ out,                  // [T][DIM] f32
    const int* __restrict__ cnt_e,
    const int* __restrict__ toklist_e,
    const float* __restrict__ wlist_e) {
    int count = *cnt_e;
    int mbase = blockIdx.x * BM;
    if (mbase >= count) return;
    int n0 = blockIdx.y * BN;

    __shared__ unsigned short As[BM * LDST];
    __shared__ unsigned short Bs[BN * LDST];

    int tid = threadIdx.x;
    int wid = tid >> 5, lane = tid & 31;
    int waveM = (wid >> 1) * 32, waveN = (wid & 1) * 32;
    int lr = lane & 15, lh = lane >> 4;

    v8f zc = {0.f,0.f,0.f,0.f,0.f,0.f,0.f,0.f};
    v8f c[2][2];
#pragma unroll
    for (int i = 0; i < 2; i++)
#pragma unroll
        for (int j = 0; j < 2; j++) c[i][j] = zc;

    for (int kb = 0; kb < HID; kb += BK) {
#pragma unroll
        for (int j = 0; j < 2; j++) {
            int chunk = tid + j * 256;
            int row = chunk >> 2, cc = (chunk & 3) * 8;
            uint4 v = *(const uint4*)(hbuf + (size_t)(mbase + row) * HID + kb + cc);
            *(uint4*)(&As[row * LDST + cc]) = v;
        }
        {
            int nn = tid >> 2, cc = (tid & 3) * 8;
            uint4 v = *(const uint4*)(W2t + (size_t)(n0 + nn) * HID + kb + cc);
            *(uint4*)(&Bs[nn * LDST + cc]) = v;
        }
        __syncthreads();

        Frag a[2], b[2];
#pragma unroll
        for (int tm = 0; tm < 2; tm++) {
            int base = (waveM + tm * 16 + lr) * LDST + lh * 16;
            a[tm].q[0] = *(const uint4*)(&As[base]);
            a[tm].q[1] = *(const uint4*)(&As[base + 8]);
        }
#pragma unroll
        for (int tn = 0; tn < 2; tn++) {
            int base = (waveN + tn * 16 + lr) * LDST + lh * 16;
            b[tn].q[0] = *(const uint4*)(&Bs[base]);
            b[tn].q[1] = *(const uint4*)(&Bs[base + 8]);
        }
#pragma unroll
        for (int tm = 0; tm < 2; tm++)
#pragma unroll
            for (int tn = 0; tn < 2; tn++)
                c[tm][tn] = wmma_bf16(a[tm].v, b[tn].v, c[tm][tn]);
        __syncthreads();
    }

    // scatter with combine weight (sequential expert launches -> no races)
#pragma unroll
    for (int tm = 0; tm < 2; tm++)
#pragma unroll
        for (int j = 0; j < 8; j++) {
            int gm = mbase + waveM + tm * 16 + j + 8 * lh;
            if (gm >= count) continue;
            int   t = toklist_e[gm];
            float w = wlist_e[gm];
#pragma unroll
            for (int tn = 0; tn < 2; tn++) {
                int n = n0 + waveN + tn * 16 + lr;
                size_t oi = (size_t)t * DIM + n;
                out[oi] += w * c[tm][tn][j];
            }
        }
}

// ---------------------------------------------------------------------------
// Launch
// ---------------------------------------------------------------------------
extern "C" void kernel_launch(void* const* d_in, const int* in_sizes, int n_in,
                              void* d_out, int out_size, void* d_ws, size_t ws_size,
                              hipStream_t stream) {
    (void)in_sizes; (void)n_in; (void)out_size; (void)ws_size;
    const float* x  = (const float*)d_in[0];
    const float* Wg = (const float*)d_in[1];
    const float* W1 = (const float*)d_in[2];
    const float* W3 = (const float*)d_in[3];
    const float* W2 = (const float*)d_in[4];
    float* out = (float*)d_out;

    // workspace carve (~126 MB)
    char* w = (char*)d_ws;
    unsigned short* xb  = (unsigned short*)w; w += (size_t)T_TOK * DIM * 2;
    unsigned short* W1t = (unsigned short*)w; w += (size_t)NEXP * HID * DIM * 2;
    unsigned short* W3t = (unsigned short*)w; w += (size_t)NEXP * HID * DIM * 2;
    unsigned short* W2t = (unsigned short*)w; w += (size_t)NEXP * DIM * HID * 2;
    unsigned short* hb  = (unsigned short*)w; w += (size_t)T_TOK * HID * 2;
    int*   cnt     = (int*)w;   w += 64;
    int*   toklist = (int*)w;   w += (size_t)NEXP * T_TOK * 4;
    float* wlist   = (float*)w; w += (size_t)NEXP * T_TOK * 4;

    init_counts<<<1, 32, 0, stream>>>(cnt);
    {
        int n4 = T_TOK * DIM / 4;
        zero_f32<<<(n4 + 255) / 256, 256, 0, stream>>>(out, n4);
        convert_x_bf16<<<(n4 + 255) / 256, 256, 0, stream>>>(x, xb, n4);
    }
    // weight convert + transpose to [N][K] bf16 (one-time, bandwidth-bound)
    transpose_to_bf16<<<dim3(HID / 32, DIM / 32, NEXP), dim3(32, 8), 0, stream>>>(
        W1, W1t, DIM, HID);
    transpose_to_bf16<<<dim3(HID / 32, DIM / 32, NEXP), dim3(32, 8), 0, stream>>>(
        W3, W3t, DIM, HID);
    transpose_to_bf16<<<dim3(DIM / 32, HID / 32, NEXP), dim3(32, 8), 0, stream>>>(
        W2, W2t, HID, DIM);

    moe_gate<<<T_TOK / 8, 256, 0, stream>>>(x, Wg, cnt, toklist, wlist);

    for (int e = 0; e < NEXP; e++) {
        const unsigned short* w1e = W1t + (size_t)e * HID * DIM;
        const unsigned short* w3e = W3t + (size_t)e * HID * DIM;
        const unsigned short* w2e = W2t + (size_t)e * DIM * HID;
        moe_gemm1<<<dim3(T_TOK / BM, HID / BN), 256, 0, stream>>>(
            xb, w1e, w3e, hb, cnt + e, toklist + e * T_TOK);
        moe_gemm2<<<dim3(T_TOK / BM, DIM / BN), 256, 0, stream>>>(
            hb, w2e, out, cnt + e, toklist + e * T_TOK, wlist + e * T_TOK);
    }
}
